// dVAE_45028437131796
// MI455X (gfx1250) — matrix operand order repaired and checked by reading, not compiled, not run on hardware
//
#include <hip/hip_runtime.h>
#include <hip/hip_bf16.h>

// ---------------------------------------------------------------------------
// dVAE forward for MI455X (gfx1250, wave32, WMMA).
// Decoder (62 GFLOP of GEMMs) fully fused per 64-row tile of the 46656-code
// book using v_wmma_f32_16x16x32_f16 with f32 accumulation; BN-eval folded
// into per-channel scale/shift. Register-blocked 4xM per wave: each B
// fragment (global/L2) is reused by 4 independent WMMA accumulator chains.
// Encoder (tiny, train-mode BN) on VALU.
// ---------------------------------------------------------------------------

#define EPS_C 1e-05f
#define ENTROPY_SCALE_C 0.1f
#define NTOT 46656            // 6^6
#define OBS_SIZE (64 * NTOT)
#define DIST_OFF OBS_SIZE     // 64*36 floats
#define LOSS_OFF (OBS_SIZE + 64 * 36)

typedef __attribute__((ext_vector_type(16))) _Float16 v16h;
typedef __attribute__((ext_vector_type(8)))  float    v8f;
typedef __attribute__((ext_vector_type(4)))  unsigned int u32x4;

union AB16 { u32x4 q[2]; v16h v; };

// ---- workspace layout (bytes, 256-aligned sections) -----------------------
static constexpr size_t OFF_W1  = 0;         // dfc1_w f16   1024*192
static constexpr size_t OFF_W2  = 393216;    // dfc2_w f16   256*1024
static constexpr size_t OFF_W3  = 917504;    // dct1_w^T f16 784*256
static constexpr size_t OFF_XH  = 1318912;   // x f16 padded 64*160
static constexpr size_t OFF_EMB = 1339392;   // embeds f16   6*6*32
static constexpr size_t OFF_S1  = 1341696;   // scale1 f32 1024
static constexpr size_t OFF_T1  = 1345792;   // shift1 f32 1024
static constexpr size_t OFF_S2  = 1349888;   // scale2 f32 256
static constexpr size_t OFF_T2  = 1350912;   // shift2 f32 256
static constexpr size_t OFF_S3  = 1351936;   // scale3 f32 16
static constexpr size_t OFF_T3  = 1352192;   // shift3 f32 16
static constexpr size_t OFF_XSQ = 1352448;   // Xsq f32 64
static constexpr size_t OFF_E1  = 1352704;   // encoder e1 scratch f32 64*1024
// total needed: 1614848 bytes (~1.6 MB) of d_ws

// ---------------------------------------------------------------------------
// WMMA fragment loaders (layouts per CDNA5 ISA 7.12.2, wave32)
// ---------------------------------------------------------------------------
static __device__ __forceinline__ v16h load_frag_a(const _Float16* A, int ld,
                                                   int m0, int k0, int lane) {
  // A 16x32 f16: lanes 0-15 row M=lane, K 0..7 & 16..23; lanes 16-31 K 8..15 & 24..31
  int row = m0 + (lane & 15);
  int kk  = k0 + ((lane & 16) >> 1);
  AB16 u;
  u.q[0] = *(const u32x4*)(A + row * ld + kk);
  u.q[1] = *(const u32x4*)(A + row * ld + kk + 16);
  return u.v;
}

static __device__ __forceinline__ v16h load_frag_b(const _Float16* W, int ld,
                                                   int n0, int k0, int lane) {
  // B 32x16 f16 from row-major W[out][in]: lane = column; lanes 0-15 K 0..7 & 16..23; lanes 16-31 K 8..15 & 24..31
  int col = n0 + (lane & 15);
  int kk  = k0 + ((lane & 16) >> 1);
  AB16 u;
  u.q[0] = *(const u32x4*)(W + col * ld + kk);
  u.q[1] = *(const u32x4*)(W + col * ld + kk + 16);
  return u.v;
}

// One fused GEMM stage: D = relu(scale * (A@W^T) + shift), f16 result to LDS.
// M = 64 rows fixed (4 register-blocked 16-row tiles per wave), NT = #16-col
// tiles, KS = #32-deep K steps. Each B fragment feeds 4 independent WMMA
// accumulator chains (no D->A/B hazard between them).
template <int NT, int KS, int CHDIV>
static __device__ __forceinline__ void gemm_stage(
    const _Float16* Alds, int ldA, const _Float16* Wg, int ldW,
    _Float16* Dlds, int ldD, const float* scale, const float* shift,
    int wave, int lane) {
  const v8f zero = {0.f, 0.f, 0.f, 0.f, 0.f, 0.f, 0.f, 0.f};
  for (int nt = wave; nt < NT; nt += 8) {
    int ni = nt << 4;
    v8f acc0 = zero, acc1 = zero, acc2 = zero, acc3 = zero;
    for (int ks = 0; ks < KS; ++ks) {
      int k0 = ks * 32;
      v16h b  = load_frag_b(Wg, ldW, ni, k0, lane);
      v16h a0 = load_frag_a(Alds, ldA, 0,  k0, lane);
      v16h a1 = load_frag_a(Alds, ldA, 16, k0, lane);
      v16h a2 = load_frag_a(Alds, ldA, 32, k0, lane);
      v16h a3 = load_frag_a(Alds, ldA, 48, k0, lane);
      acc0 = __builtin_amdgcn_wmma_f32_16x16x32_f16(false, a0, false, b,
                                                    (short)0, acc0, false, false);
      acc1 = __builtin_amdgcn_wmma_f32_16x16x32_f16(false, a1, false, b,
                                                    (short)0, acc1, false, false);
      acc2 = __builtin_amdgcn_wmma_f32_16x16x32_f16(false, a2, false, b,
                                                    (short)0, acc2, false, false);
      acc3 = __builtin_amdgcn_wmma_f32_16x16x32_f16(false, a3, false, b,
                                                    (short)0, acc3, false, false);
    }
    int col = ni + (lane & 15);
    int ch  = (CHDIV == 1) ? col : (col / CHDIV);
    float s = scale[ch], sh = shift[ch];
    int mb  = (lane & 16) >> 1;
    v8f accs[4] = {acc0, acc1, acc2, acc3};
#pragma unroll
    for (int mi = 0; mi < 4; ++mi) {
#pragma unroll
      for (int r = 0; r < 8; ++r) {
        float y = s * accs[mi][r] + sh;
        y = y > 0.f ? y : 0.f;
        Dlds[(mi * 16 + mb + r) * ldD + col] = (_Float16)y;
      }
    }
  }
}

// ---------------------------------------------------------------------------
// Prep: weights -> f16 (dct1_w transposed), BN-eval scale/shift, X f16 + Xsq
// ---------------------------------------------------------------------------
__global__ void prep_kernel(
    const float* x, const float* embeds,
    const float* dfc1_w, const float* dfc1_b,
    const float* dbn1_g, const float* dbn1_b, const float* dbn1_rm, const float* dbn1_rv,
    const float* dfc2_w, const float* dfc2_b,
    const float* dbn2_g, const float* dbn2_b, const float* dbn2_rm, const float* dbn2_rv,
    const float* dct1_w, const float* dct1_b,
    const float* dbn3_g, const float* dbn3_b, const float* dbn3_rm, const float* dbn3_rv,
    char* ws) {
  int tid = blockIdx.x * blockDim.x + threadIdx.x;
  int np  = gridDim.x * blockDim.x;
  _Float16* W1h  = (_Float16*)(ws + OFF_W1);
  _Float16* W2h  = (_Float16*)(ws + OFF_W2);
  _Float16* W3h  = (_Float16*)(ws + OFF_W3);
  _Float16* XH   = (_Float16*)(ws + OFF_XH);
  _Float16* embH = (_Float16*)(ws + OFF_EMB);
  float* s1 = (float*)(ws + OFF_S1);  float* t1 = (float*)(ws + OFF_T1);
  float* s2 = (float*)(ws + OFF_S2);  float* t2 = (float*)(ws + OFF_T2);
  float* s3 = (float*)(ws + OFF_S3);  float* t3 = (float*)(ws + OFF_T3);
  float* Xsq = (float*)(ws + OFF_XSQ);

  for (int i = tid; i < 1024 * 192; i += np) W1h[i] = (_Float16)dfc1_w[i];
  for (int i = tid; i < 256 * 1024; i += np) W2h[i] = (_Float16)dfc2_w[i];
  for (int i = tid; i < 256 * 784; i += np) {   // transpose (256,784)->(784,256)
    int ic = i / 784, o = i - ic * 784;
    W3h[o * 256 + ic] = (_Float16)dct1_w[i];
  }
  for (int i = tid; i < 6 * 6 * 32; i += np) embH[i] = (_Float16)embeds[i];
  for (int i = tid; i < 64 * 160; i += np) {    // pad K 147 -> 160 with zeros
    int b = i / 160, k = i - b * 160;
    XH[i] = (k < 147) ? (_Float16)x[b * 147 + k] : (_Float16)0.f;
  }
  for (int i = tid; i < 64; i += np) {
    float s = 0.f; const float* xp = x + i * 147;
    for (int k = 0; k < 147; ++k) s += xp[k] * xp[k];
    Xsq[i] = s;
  }
  for (int i = tid; i < 1024; i += np) {
    float sc = dbn1_g[i] * rsqrtf(dbn1_rv[i] + EPS_C);
    s1[i] = sc; t1[i] = sc * (dfc1_b[i] - dbn1_rm[i]) + dbn1_b[i];
  }
  for (int i = tid; i < 256; i += np) {
    float sc = dbn2_g[i] * rsqrtf(dbn2_rv[i] + EPS_C);
    s2[i] = sc; t2[i] = sc * (dfc2_b[i] - dbn2_rm[i]) + dbn2_b[i];
  }
  for (int i = tid; i < 16; i += np) {
    float sc = dbn3_g[i] * rsqrtf(dbn3_rv[i] + EPS_C);
    s3[i] = sc; t3[i] = sc * (dct1_b[i] - dbn3_rm[i]) + dbn3_b[i];
  }
}

// ---------------------------------------------------------------------------
// Encoder: conv-as-GEMM + train-BN + fc1 + train-BN + logits + softmax/entropy.
// Tiny (~40 MFLOP); one 1024-thread workgroup on VALU.
// ---------------------------------------------------------------------------
__global__ void __launch_bounds__(1024) encoder_kernel(
    const float* x, const float* conv_w, const float* conv_b,
    const float* ebn2_g, const float* ebn2_b,
    const float* fc1_w, const float* fc1_b,
    const float* ebn1_g, const float* ebn1_b,
    const float* fc2_w, const float* fc2_b,
    char* ws, float* dout) {
  extern __shared__ char smem[];
  float* eL      = (float*)smem;                    // 64*256
  float* logitsL = (float*)(smem + 65536);          // 64*36
  float* lossL   = (float*)(smem + 65536 + 9216);   // 64
  float* e1      = (float*)(ws + OFF_E1);           // 64*1024 global scratch
  int tid = threadIdx.x;

  // conv: e0[b,o] = x[b,:147] . conv_w[o,:147] + conv_b[o]
  for (int idx = tid; idx < 64 * 256; idx += 1024) {
    int b = idx >> 8, o = idx & 255;
    const float* xp = x + b * 147;
    const float* wp = conv_w + o * 147;
    float s = conv_b[o];
    for (int k = 0; k < 147; ++k) s += xp[k] * wp[k];
    eL[idx] = s;
  }
  __syncthreads();
  // train-mode BN over batch (per channel of 256) + relu, in place
  if (tid < 256) {
    float m = 0.f;
    for (int b = 0; b < 64; ++b) m += eL[b * 256 + tid];
    m *= (1.f / 64.f);
    float v = 0.f;
    for (int b = 0; b < 64; ++b) { float d = eL[b * 256 + tid] - m; v += d * d; }
    v *= (1.f / 64.f);
    float inv = rsqrtf(v + EPS_C);
    float g = ebn2_g[tid], bb = ebn2_b[tid];
    for (int b = 0; b < 64; ++b) {
      float y = g * (eL[b * 256 + tid] - m) * inv + bb;
      eL[b * 256 + tid] = y > 0.f ? y : 0.f;
    }
  }
  __syncthreads();
  // fc1 -> e1 (global scratch)
  for (int idx = tid; idx < 64 * 1024; idx += 1024) {
    int b = idx >> 10, j = idx & 1023;
    const float* ep = eL + b * 256;
    const float* wp = fc1_w + j * 256;
    float s = fc1_b[j];
    for (int k = 0; k < 256; ++k) s += ep[k] * wp[k];
    e1[idx] = s;
  }
  __threadfence();
  __syncthreads();
  // train-mode BN over batch (per channel of 1024) + relu, in place
  {
    int j = tid;
    float m = 0.f;
    for (int b = 0; b < 64; ++b) m += e1[b * 1024 + j];
    m *= (1.f / 64.f);
    float v = 0.f;
    for (int b = 0; b < 64; ++b) { float d = e1[b * 1024 + j] - m; v += d * d; }
    v *= (1.f / 64.f);
    float inv = rsqrtf(v + EPS_C);
    float g = ebn1_g[j], bb = ebn1_b[j];
    for (int b = 0; b < 64; ++b) {
      float y = g * (e1[b * 1024 + j] - m) * inv + bb;
      e1[b * 1024 + j] = y > 0.f ? y : 0.f;
    }
  }
  __threadfence();
  __syncthreads();
  // logits = e1 @ fc2_w^T + fc2_b  (64 x 36)
  for (int idx = tid; idx < 64 * 36; idx += 1024) {
    int b = idx / 36, q = idx - b * 36;
    const float* ep = e1 + b * 1024;
    const float* wp = fc2_w + q * 1024;
    float s = fc2_b[q];
    for (int k = 0; k < 1024; ++k) s += ep[k] * wp[k];
    logitsL[idx] = s;
  }
  if (tid < 64) lossL[tid] = 0.f;
  __syncthreads();
  // softmax over K=6 within each (b,v) group, entropy loss
  for (int g = tid; g < 64 * 6; g += 1024) {
    int b = g / 6, v = g - b * 6;
    float* lp = logitsL + b * 36 + v * 6;
    float mx = lp[0];
    for (int k = 1; k < 6; ++k) mx = fmaxf(mx, lp[k]);
    float ex[6], sum = 0.f;
    for (int k = 0; k < 6; ++k) { ex[k] = expf(lp[k] - mx); sum += ex[k]; }
    float invs = 1.f / sum, ent = 0.f;
    for (int k = 0; k < 6; ++k) {
      float p = ex[k] * invs;
      dout[DIST_OFF + b * 36 + v * 6 + k] = p;
      ent += p * logf(p + 1e-10f);
    }
    atomicAdd(&lossL[b], ENTROPY_SCALE_C * ent);
  }
  __syncthreads();
  if (tid < 64) dout[LOSS_OFF + tid] = lossL[tid];
}

// ---------------------------------------------------------------------------
// Fused decoder + observation kernel: 729 blocks x 64 codebook rows each.
// LDS (dynamic, 160 KB) is time-multiplexed across stages.
// ---------------------------------------------------------------------------
__global__ void __launch_bounds__(256) decoder_kernel(const char* ws,
                                                      const float* dct2_w,
                                                      const float* dct2_b,
                                                      float* out) {
  extern __shared__ char smem[];
  _Float16* h1H  = (_Float16*)(smem);            // [  0,131072) 64x1024
  _Float16* zH   = (_Float16*)(smem + 131072);   // [131072,155648) 64x192
  _Float16* h2H  = (_Float16*)(smem + 131072);   // 64x256 (after zH dead)
  _Float16* h3H  = (_Float16*)(smem);            // 64x784 (after h1H dead)
  _Float16* EH   = (_Float16*)(smem + 131072);   // 64x160 (after h2H dead)
  float*    Esq  = (float*)(smem + 151552);      // 64
  float*    obsT = (float*)(smem);               // 64x64 (after h3H dead)
  __shared__ float d2wL[48];
  __shared__ float d2bL[3];

  const _Float16* W1h  = (const _Float16*)(ws + OFF_W1);
  const _Float16* W2h  = (const _Float16*)(ws + OFF_W2);
  const _Float16* W3h  = (const _Float16*)(ws + OFF_W3);
  const _Float16* XHg  = (const _Float16*)(ws + OFF_XH);
  const _Float16* embH = (const _Float16*)(ws + OFF_EMB);
  const float* s1 = (const float*)(ws + OFF_S1);
  const float* t1 = (const float*)(ws + OFF_T1);
  const float* s2 = (const float*)(ws + OFF_S2);
  const float* t2 = (const float*)(ws + OFF_T2);
  const float* s3 = (const float*)(ws + OFF_S3);
  const float* t3 = (const float*)(ws + OFF_T3);
  const float* XsqW = (const float*)(ws + OFF_XSQ);

  int tid  = threadIdx.x;
  int lane = tid & 31;
  int wave = tid >> 5;
  int n0   = blockIdx.x * 64;

  if (tid < 48) d2wL[tid] = dct2_w[tid];
  if (tid < 3)  d2bL[tid] = dct2_b[tid];

  // --- build z tile: z[m, v*32+d] = embeds[v, digit_v(n0+m), d] ---
  const int pw[6] = {7776, 1296, 216, 36, 6, 1};
  for (int idx = tid; idx < 64 * 192; idx += 256) {
    int m = idx / 192, j = idx - m * 192;
    int v = j >> 5, d = j & 31;
    int n = n0 + m;
    int dig = (n / pw[v]) % 6;
    zH[m * 192 + j] = embH[(v * 6 + dig) * 32 + d];
  }
  __syncthreads();

  // --- h1 = relu(bn(z @ dfc1^T)) : M64 x N1024, K192 ---
  gemm_stage<64, 6, 1>(zH, 192, W1h, 192, h1H, 1024, s1, t1, wave, lane);
  __syncthreads();
  // --- h2 = relu(bn(h1 @ dfc2^T)) : M64 x N256, K1024 ---
  gemm_stage<16, 32, 1>(h1H, 1024, W2h, 1024, h2H, 256, s2, t2, wave, lane);
  __syncthreads();
  // --- h3 = relu(bn(h2 @ dct1)) : M64 x N784 (o*49+hw), K256; BN channel = col/49
  gemm_stage<49, 8, 49>(h2H, 256, W3h, 256, h3H, 784, s3, t3, wave, lane);
  __syncthreads();

  // --- E[m, c*49+hw] = sum_o h3[m, o*49+hw] * dct2_w[o,c] + dct2_b[c]; pad->160
  for (int idx = tid; idx < 64 * 160; idx += 256) {
    int m = idx / 160, col = idx - m * 160;
    float val = 0.f;
    if (col < 147) {
      int c = col / 49, hw = col - c * 49;
      float s = d2bL[c];
      const _Float16* hp = h3H + m * 784 + hw;
#pragma unroll
      for (int o2 = 0; o2 < 16; ++o2) s += (float)hp[o2 * 49] * d2wL[o2 * 3 + c];
      val = s;
    }
    EH[idx] = (_Float16)val;
  }
  __syncthreads();

  if (tid < 64) {
    float s = 0.f;
    const _Float16* ep = EH + tid * 160;
    for (int k = 0; k < 147; ++k) { float e = (float)ep[k]; s += e * e; }
    Esq[tid] = s;
  }
  __syncthreads();

  // --- obsT[b][m] = E[m,:] . X[b,:]  (64x64 tile, K=160 incl. zero pad).
  // Register-blocked like the main stages: 4 waves each own one 16-col b-tile,
  // reuse its B fragment across the 4 M tiles (4 independent WMMA chains).
  if (wave < 4) {
    int ni = wave << 4;
    const v8f zero = {0.f, 0.f, 0.f, 0.f, 0.f, 0.f, 0.f, 0.f};
    v8f acc0 = zero, acc1 = zero, acc2 = zero, acc3 = zero;
    for (int ks = 0; ks < 5; ++ks) {
      int k0 = ks * 32;
      v16h b  = load_frag_b(XHg, 160, ni, k0, lane);
      v16h a0 = load_frag_a(EH, 160, 0,  k0, lane);
      v16h a1 = load_frag_a(EH, 160, 16, k0, lane);
      v16h a2 = load_frag_a(EH, 160, 32, k0, lane);
      v16h a3 = load_frag_a(EH, 160, 48, k0, lane);
      acc0 = __builtin_amdgcn_wmma_f32_16x16x32_f16(false, a0, false, b,
                                                    (short)0, acc0, false, false);
      acc1 = __builtin_amdgcn_wmma_f32_16x16x32_f16(false, a1, false, b,
                                                    (short)0, acc1, false, false);
      acc2 = __builtin_amdgcn_wmma_f32_16x16x32_f16(false, a2, false, b,
                                                    (short)0, acc2, false, false);
      acc3 = __builtin_amdgcn_wmma_f32_16x16x32_f16(false, a3, false, b,
                                                    (short)0, acc3, false, false);
    }
    int bcol = ni + (lane & 15);
    int mb   = (lane & 16) >> 1;
    v8f accs[4] = {acc0, acc1, acc2, acc3};
#pragma unroll
    for (int mi = 0; mi < 4; ++mi)
#pragma unroll
      for (int r = 0; r < 8; ++r)
        obsT[bcol * 64 + mi * 16 + mb + r] = accs[mi][r];
  }
  __syncthreads();

  // --- obs[b, n0+j] = -0.5*(Esq[j] - 2*XE[b,j] + Xsq[b]) ---
  for (int idx = tid; idx < 64 * 64; idx += 256) {
    int b = idx >> 6, j = idx & 63;
    out[(size_t)b * NTOT + n0 + j] =
        -0.5f * (Esq[j] - 2.f * obsT[b * 64 + j] + XsqW[b]);
  }
}

// ---------------------------------------------------------------------------
extern "C" void kernel_launch(void* const* d_in, const int* in_sizes, int n_in,
                              void* d_out, int out_size, void* d_ws, size_t ws_size,
                              hipStream_t stream) {
  (void)in_sizes; (void)n_in; (void)out_size; (void)ws_size;
  const float* x      = (const float*)d_in[0];
  const float* conv_w = (const float*)d_in[1];
  const float* conv_b = (const float*)d_in[2];
  const float* ebn2_g = (const float*)d_in[3];
  const float* ebn2_b = (const float*)d_in[4];
  const float* fc1_w  = (const float*)d_in[5];
  const float* fc1_b  = (const float*)d_in[6];
  const float* ebn1_g = (const float*)d_in[7];
  const float* ebn1_b = (const float*)d_in[8];
  const float* fc2_w  = (const float*)d_in[9];
  const float* fc2_b  = (const float*)d_in[10];
  const float* embeds = (const float*)d_in[11];
  const float* dfc1_w = (const float*)d_in[12];
  const float* dfc1_b = (const float*)d_in[13];
  const float* dbn1_g = (const float*)d_in[14];
  const float* dbn1_b = (const float*)d_in[15];
  const float* dbn1_rm = (const float*)d_in[16];
  const float* dbn1_rv = (const float*)d_in[17];
  const float* dfc2_w = (const float*)d_in[18];
  const float* dfc2_b = (const float*)d_in[19];
  const float* dbn2_g = (const float*)d_in[20];
  const float* dbn2_b = (const float*)d_in[21];
  const float* dbn2_rm = (const float*)d_in[22];
  const float* dbn2_rv = (const float*)d_in[23];
  const float* dct1_w = (const float*)d_in[24];
  const float* dct1_b = (const float*)d_in[25];
  const float* dbn3_g = (const float*)d_in[26];
  const float* dbn3_b = (const float*)d_in[27];
  const float* dbn3_rm = (const float*)d_in[28];
  const float* dbn3_rv = (const float*)d_in[29];
  const float* dct2_w = (const float*)d_in[30];
  const float* dct2_b = (const float*)d_in[31];

  char*  ws  = (char*)d_ws;
  float* out = (float*)d_out;

  prep_kernel<<<512, 256, 0, stream>>>(
      x, embeds, dfc1_w, dfc1_b, dbn1_g, dbn1_b, dbn1_rm, dbn1_rv,
      dfc2_w, dfc2_b, dbn2_g, dbn2_b, dbn2_rm, dbn2_rv,
      dct1_w, dct1_b, dbn3_g, dbn3_b, dbn3_rm, dbn3_rv, ws);

  encoder_kernel<<<1, 1024, 75008, stream>>>(
      x, conv_w, conv_b, ebn2_g, ebn2_b, fc1_w, fc1_b,
      ebn1_g, ebn1_b, fc2_w, fc2_b, ws, out);

  decoder_kernel<<<729, 256, 163840, stream>>>(ws, dct2_w, dct2_b, out);
}
